// LePEAttention_91122026152555
// MI455X (gfx1250) — matrix operand order repaired and checked by reading, not compiled, not run on hardware
//
#include <hip/hip_runtime.h>
#include <hip/hip_bf16.h>
#include <stdint.h>

// ---------------------------------------------------------------------------
// LePE windowed attention (CSWin style) for MI455X / gfx1250.
// All GEMMs use v_wmma_f32_16x16x32_bf16 (bf16 in, f32 accumulate).
// ---------------------------------------------------------------------------

typedef __attribute__((ext_vector_type(16))) __bf16 v16bf;
typedef __attribute__((ext_vector_type(8)))  float  v8f;

#define HSP   56
#define WSP   7
#define LWIN  392     // 56*7
#define LPAD  400     // padded L for 16-row tiling
#define LPAD2 416     // padded L for K=32 tiling in attn@V
#define NRT   25      // 400/16 row tiles
#define NCT   25      // 400/16 col tiles (real)
#define NCTP  28      // padded col tiles -> uniform 7 tiles per wave
#define NKT   13      // 416/32 K tiles for attn@V
#define NHEADS 8
#define HD    32
#define CCH   256
#define NWIN  64      // 8 batch * 8 windows

__device__ __forceinline__ unsigned short f2bf(float f) {
  union { float f; uint32_t u; } v; v.f = f;
  uint32_t u = v.u;
  uint32_t r = u + 0x7FFFu + ((u >> 16) & 1u);   // round-to-nearest-even
  return (unsigned short)(r >> 16);
}
__device__ __forceinline__ float bf2f(unsigned short h) {
  union { uint32_t u; float f; } v; v.u = ((uint32_t)h) << 16;
  return v.f;
}
__device__ __forceinline__ v8f v8zero() {
  v8f z = {0.f,0.f,0.f,0.f,0.f,0.f,0.f,0.f}; return z;
}

// A-matrix (16xK, 16-bit) fragment swizzle: element (m, kk) -> ushort index
// inside a 32-lane x 16-ushort tile. Lanes 0-15: M=lane, K in {0-7,16-23};
// lanes 16-31: M=lane-16, K in {8-15,24-31}.  (ISA 7.12.2)
__device__ __forceinline__ int a_swz(int m, int kk) {
  int hi = (kk >> 3) & 1;
  int j  = (kk < 16) ? (kk - 8 * hi) : (kk - 8 - 8 * hi);
  return (m + hi * 16) * 16 + j;
}
// B-matrix (32x16, 16-bit) fragment swizzle: element (kk, n).
// Lanes 0-15: N=lane, K in {0-7,16-23}; lanes 16-31: N=lane-16, K in {8-15,24-31}.
__device__ __forceinline__ int b_swz(int n, int kk) {
  int hi = (kk >> 3) & 1;
  int j  = (kk < 16) ? (kk - 8 * hi) : (kk - 8 - 8 * hi);
  return ((n & 15) + hi * 16) * 16 + j;
}

// ---------------------------------------------------------------------------
// Kernel 1: t = v + dwconv3x3(v), per window (zero-pad at window borders),
// output bf16 into t_ws[win][c][LPAD].
// ---------------------------------------------------------------------------
__global__ __launch_bounds__(64)
void prep_t_kernel(const float* __restrict__ temp,
                   const float* __restrict__ w_dw,
                   const float* __restrict__ b_dw,
                   unsigned short* __restrict__ t_ws) {
  __shared__ float vt[58 * 9];                 // (56+2) x (7+2) halo tile
  int win = blockIdx.x >> 8;
  int c   = blockIdx.x & 255;
  int b   = win >> 3, ww = win & 7;
  int tid = threadIdx.x;

  const float* vsrc = temp + (((size_t)(b * 3 + 2) * CCH + c) * 3136);
  for (int i = tid; i < 58 * 9; i += 64) vt[i] = 0.f;
  __syncthreads();
  for (int i = tid; i < LWIN; i += 64) {
    int y = i / 7, x = i - y * 7;
    vt[(y + 1) * 9 + (x + 1)] = vsrc[y * 56 + ww * 7 + x];
  }
  __syncthreads();

  float w[9];
#pragma unroll
  for (int t = 0; t < 9; ++t) w[t] = w_dw[c * 9 + t];
  float bb = b_dw[c];

  unsigned short* dst = t_ws + ((size_t)win * CCH + c) * LPAD;
  for (int i = tid; i < LWIN; i += 64) {
    int y = i / 7, x = i - y * 7;
    float acc = bb;
#pragma unroll
    for (int dy = 0; dy < 3; ++dy)
#pragma unroll
      for (int dx = 0; dx < 3; ++dx)
        acc = fmaf(w[dy * 3 + dx], vt[(y + dy) * 9 + (x + dx)], acc);
    dst[i] = f2bf(vt[(y + 1) * 9 + (x + 1)] + acc);
  }
}

// ---------------------------------------------------------------------------
// Kernel 2: rpe = conv3x3(t, w_pw) + b_pw as 9 shifted GEMM taps.
// Grid: 64 windows x 16 out-channel tiles. Block: 128 (4 waves).
// M=16 (cout tile), N padded to 28 tiles (7 per wave, uniform), K=9*256 via
// 72 staged steps; 7 B-fragments preloaded -> 7 back-to-back WMMAs.
// ---------------------------------------------------------------------------
__global__ __launch_bounds__(128)
void pwconv_kernel(const unsigned short* __restrict__ t_ws,
                   const float* __restrict__ w_pw,
                   const float* __restrict__ b_pw,
                   unsigned short* __restrict__ rpe_ws) {
  __shared__ __align__(32) unsigned short alds[32 * 16];             // 1 KB
  __shared__ __align__(32) unsigned short blds[NCTP * 32 * 16];      // 28.7 KB

  int win = __builtin_amdgcn_readfirstlane(blockIdx.x >> 4);
  int mt  = __builtin_amdgcn_readfirstlane(blockIdx.x & 15);
  int tid = threadIdx.x;
  int wave = __builtin_amdgcn_readfirstlane(tid >> 5);   // SGPR: scalar guards
  int lane = tid & 31;

  // Zero the 3 padding tiles once (never restaged; A*0 accumulates nothing).
  for (int i = tid; i < (NCTP - NCT) * 32 * 16; i += 128)
    blds[NCT * 32 * 16 + i] = 0;

  v8f acc[7];
#pragma unroll
  for (int i = 0; i < 7; ++i) acc[i] = v8zero();

  const unsigned short* tw = t_ws + (size_t)win * CCH * LPAD;

  for (int tap = 0; tap < 9; ++tap) {
    int dy = tap / 3 - 1, dx = tap % 3 - 1;
    for (int kt = 0; kt < 8; ++kt) {
      __syncthreads();
      // Stage A: 4 weight loads in flight, then 4 convert+stores.
      {
        float wv[4];
#pragma unroll
        for (int u = 0; u < 4; ++u) {
          int i = tid + u * 128;
          int m = i >> 5, kk = i & 31;
          wv[u] = w_pw[(((size_t)(mt * 16 + m) * CCH + kt * 32 + kk) * 9) + tap];
        }
#pragma unroll
        for (int u = 0; u < 4; ++u) {
          int i = tid + u * 128;
          int m = i >> 5, kk = i & 31;
          alds[a_swz(m, kk)] = f2bf(wv[u]);
        }
      }
      // One speculative prefetch per thread toward the next k-panel.
      if (kt < 7)
        __builtin_prefetch(tw + ((size_t)((kt + 1) * 32 + (tid & 31))) * LPAD
                              + (tid >> 5) * 100, 0, 1);
      // Stage B: groups of 4 -> 4 loads outstanding, then 4 LDS stores.
      for (int g0 = 0; g0 < 100; g0 += 4) {
        unsigned short bv[4];
        int di[4];
#pragma unroll
        for (int u = 0; u < 4; ++u) {
          int i = tid + (g0 + u) * 128;
          int kk = i / LPAD;
          int n  = i - kk * LPAD;
          unsigned short v = 0;
          if (n < LWIN) {
            int y = n / 7 + dy, x = n - (n / 7) * 7 + dx;
            if (y >= 0 && y < HSP && x >= 0 && x < WSP)
              v = tw[((size_t)(kt * 32 + kk)) * LPAD + y * 7 + x];
          }
          bv[u] = v;
          di[u] = (n >> 4) * 512 + b_swz(n, kk);
        }
#pragma unroll
        for (int u = 0; u < 4; ++u) blds[di[u]] = bv[u];
      }
      __syncthreads();

      // Preload A + all 7 B fragments, then 7 back-to-back WMMAs.
      v16bf afrag = *(const v16bf*)(alds + lane * 16);
      v16bf bfrag[7];
#pragma unroll
      for (int i = 0; i < 7; ++i)
        bfrag[i] = *(const v16bf*)(blds + ((wave + i * 4) * 32 + lane) * 16);
#pragma unroll
      for (int i = 0; i < 7; ++i)
        acc[i] = __builtin_amdgcn_wmma_f32_16x16x32_bf16(
            false, afrag, false, bfrag[i], (short)0, acc[i], false, false);
    }
  }

  // Epilogue: C/D layout (VGPR g -> M=g or g+8, lane -> N), write bf16 rpe.
#pragma unroll
  for (int i = 0; i < 7; ++i) {
    int ct = wave + i * 4;
    if (ct < NCT) {                       // scalar guard (wave is SGPR)
#pragma unroll
      for (int g = 0; g < 8; ++g) {
        int m = g + ((lane >> 4) << 3);
        int l = ct * 16 + (lane & 15);
        if (l < LWIN) {
          int cout = mt * 16 + m;
          rpe_ws[((size_t)win * CCH + cout) * LPAD + l] =
              f2bf(acc[i][g] + b_pw[cout]);
        }
      }
    }
  }
}

// ---------------------------------------------------------------------------
// Kernel 3: windowed attention + rpe add. Grid: 64 win x 8 heads. Block: 128.
// Q/K/V staged once in fragment-native bf16 LDS; per-wave score strip + bf16
// attn strip; masked softmax with wave32 shuffles; software-pipelined
// fragment loads; output fused with rpe.
// ---------------------------------------------------------------------------
__global__ __launch_bounds__(128)
void attn_kernel(const float* __restrict__ temp,
                 const unsigned short* __restrict__ rpe_ws,
                 float* __restrict__ out) {
  __shared__ __align__(32) unsigned short qlds[NRT * 32 * 16];        // 25.6 KB
  __shared__ __align__(32) unsigned short klds[NCT * 32 * 16];        // 25.6 KB
  __shared__ __align__(32) unsigned short vlds[NKT * 2 * 32 * 16];    // 26.6 KB
  __shared__ __align__(32) unsigned short attlds[4 * NKT * 32 * 16];  // 53.2 KB
  __shared__ float slds[4 * 16 * LPAD2];                              // 106.5 KB

  int win  = __builtin_amdgcn_readfirstlane(blockIdx.x >> 3);
  int head = __builtin_amdgcn_readfirstlane(blockIdx.x & 7);
  int b = win >> 3, ww = win & 7;
  int tid = threadIdx.x;
  int wave = __builtin_amdgcn_readfirstlane(tid >> 5);   // SGPR: scalar loops
  int lane = tid & 31;
  const float scale = 0.17677669529663687f;   // 32^-0.5

  // Zero pads (rows/cols 392..415 must be exact zeros).
  for (int i = tid; i < NRT * 32 * 16; i += 128) { qlds[i] = 0; klds[i] = 0; }
  for (int i = tid; i < NKT * 2 * 32 * 16; i += 128) vlds[i] = 0;
  __syncthreads();

  const float* qsrc = temp + (((size_t)(b * 3 + 0) * CCH + head * HD) * 3136);
  const float* ksrc = temp + (((size_t)(b * 3 + 1) * CCH + head * HD) * 3136);
  const float* vsrc = temp + (((size_t)(b * 3 + 2) * CCH + head * HD) * 3136);

  for (int i = tid; i < LWIN * HD; i += 128) {
    int l = i >> 5, d = i & 31;
    int y = l / 7, x = ww * 7 + (l - y * 7);
    size_t off = (size_t)d * 3136 + y * 56 + x;
    float qv = qsrc[off];
    float kv = ksrc[off];
    float vv = vsrc[off];
    // Q as A-fragments (scaled), K as B-fragments (K^T), V as B-fragments.
    int rt = l >> 4, m = l & 15;
    qlds[rt * 512 + a_swz(m, d)] = f2bf(qv * scale);
    klds[rt * 512 + b_swz(l, d)] = f2bf(kv);             // n=l, kk=d
    int kt2 = l >> 5, kk = l & 31, nt = d >> 4;
    vlds[(kt2 * 2 + nt) * 512 + b_swz(d, kk)] = f2bf(vv);
  }
  __syncthreads();

  float* sw = slds + wave * 16 * LPAD2;
  unsigned short* aw = attlds + wave * NKT * 512;

  for (int rt = wave; rt < NRT; rt += 4) {               // scalar loop bounds
    v16bf qa = *(const v16bf*)(qlds + (rt * 32 + lane) * 16);

    // ---- scores: S = Q_tile(16x32) * K^T tiles -> f32 LDS strip ----
    // Software pipeline: next K fragment loads overlap WMMA + score stores.
    v16bf kb = *(const v16bf*)(klds + (0 * 32 + lane) * 16);
    for (int ct = 0; ct < NCT; ++ct) {
      int ctn = (ct + 1 < NCT) ? ct + 1 : ct;
      v16bf kbn = *(const v16bf*)(klds + (ctn * 32 + lane) * 16);
      v8f s = v8zero();
      s = __builtin_amdgcn_wmma_f32_16x16x32_bf16(
          false, qa, false, kb, (short)0, s, false, false);
      int nn = ct * 16 + (lane & 15);
      int mb = (lane >> 4) << 3;
#pragma unroll
      for (int g = 0; g < 8; ++g) sw[(g + mb) * LPAD2 + nn] = s[g];
      kb = kbn;
    }

    // ---- masked softmax per row, wave32 shuffle reductions ----
    for (int r = 0; r < 16; ++r) {
      float vals[13];
      float mx = -3.0e38f;
#pragma unroll
      for (int ii = 0; ii < 13; ++ii) {
        int col = lane + ii * 32;
        float v = (col < LWIN) ? sw[r * LPAD2 + col] : -3.0e38f;
        vals[ii] = v;
        mx = fmaxf(mx, v);
      }
#pragma unroll
      for (int o = 16; o >= 1; o >>= 1) mx = fmaxf(mx, __shfl_xor(mx, o, 32));
      float sum = 0.f;
#pragma unroll
      for (int ii = 0; ii < 13; ++ii) {
        int col = lane + ii * 32;
        float e = (col < LWIN) ? __expf(vals[ii] - mx) : 0.f;
        vals[ii] = e;
        sum += e;
      }
#pragma unroll
      for (int o = 16; o >= 1; o >>= 1) sum += __shfl_xor(sum, o, 32);
      float inv = 1.0f / sum;
#pragma unroll
      for (int ii = 0; ii < 13; ++ii) {
        int col = lane + ii * 32;
        int kt = col >> 5, kk = col & 31;
        aw[kt * 512 + a_swz(r, kk)] = f2bf(vals[ii] * inv);
      }
    }

    // ---- O = attn(16x416) * V(416x32): 13 K-steps, 2 N-tiles ----
    v8f o0 = v8zero(), o1 = v8zero();
    v16bf pa  = *(const v16bf*)(aw + (0 * 32 + lane) * 16);
    v16bf vb0 = *(const v16bf*)(vlds + (0 * 32 + lane) * 16);
    v16bf vb1 = *(const v16bf*)(vlds + (1 * 32 + lane) * 16);
    for (int kt = 0; kt < NKT; ++kt) {
      int ktn = (kt + 1 < NKT) ? kt + 1 : kt;
      v16bf pan  = *(const v16bf*)(aw + (ktn * 32 + lane) * 16);
      v16bf vb0n = *(const v16bf*)(vlds + ((ktn * 2 + 0) * 32 + lane) * 16);
      v16bf vb1n = *(const v16bf*)(vlds + ((ktn * 2 + 1) * 32 + lane) * 16);
      o0 = __builtin_amdgcn_wmma_f32_16x16x32_bf16(
          false, pa, false, vb0, (short)0, o0, false, false);
      o1 = __builtin_amdgcn_wmma_f32_16x16x32_bf16(
          false, pa, false, vb1, (short)0, o1, false, false);
      pa = pan; vb0 = vb0n; vb1 = vb1n;
    }

    // ---- epilogue: + rpe, write (B, H*W, C) fp32 output ----
    int mb = (lane >> 4) << 3;
    int nn = lane & 15;
#pragma unroll
    for (int g = 0; g < 8; ++g) {
      int l = rt * 16 + g + mb;
      if (l < LWIN) {
        int y = l / 7, X = ww * 7 + (l - y * 7);
        size_t obase = ((size_t)b * 3136 + y * 56 + X) * CCH;
        {
          int c = head * HD + nn;
          float r = bf2f(rpe_ws[((size_t)win * CCH + c) * LPAD + l]);
          out[obase + c] = o0[g] + r;
        }
        {
          int c = head * HD + 16 + nn;
          float r = bf2f(rpe_ws[((size_t)win * CCH + c) * LPAD + l]);
          out[obase + c] = o1[g] + r;
        }
      }
    }
  }
}

// ---------------------------------------------------------------------------
// Launch
// ---------------------------------------------------------------------------
extern "C" void kernel_launch(void* const* d_in, const int* in_sizes, int n_in,
                              void* d_out, int out_size, void* d_ws, size_t ws_size,
                              hipStream_t stream) {
  const float* temp = (const float*)d_in[0];
  const float* w_dw = (const float*)d_in[1];
  const float* b_dw = (const float*)d_in[2];
  const float* w_pw = (const float*)d_in[3];
  const float* b_pw = (const float*)d_in[4];
  float* out = (float*)d_out;

  unsigned short* t_ws   = (unsigned short*)d_ws;                    // 13.1 MB
  unsigned short* rpe_ws = t_ws + (size_t)NWIN * CCH * LPAD;         // 13.1 MB

  prep_t_kernel<<<NWIN * CCH, 64, 0, stream>>>(temp, w_dw, b_dw, t_ws);
  pwconv_kernel<<<NWIN * 16, 128, 0, stream>>>(t_ws, w_pw, b_pw, rpe_ws);
  attn_kernel<<<NWIN * NHEADS, 128, 0, stream>>>(temp, rpe_ws, out);
}
